// AttentionModule_43885975830652
// MI455X (gfx1250) — matrix operand order repaired and checked by reading, not compiled, not run on hardware
//
#include <hip/hip_runtime.h>
#include <math.h>

// ---------------------------------------------------------------------------
// Problem constants (B=8, N=2048, C=512 from the reference)
// ---------------------------------------------------------------------------
#define C_DIM 512
#define N_TOK 2048
#define B_DIM 8
#define M_TOT (B_DIM * N_TOK) // 16384 total tokens

typedef _Float16 h8_t  __attribute__((ext_vector_type(8)));
typedef _Float16 v16h  __attribute__((ext_vector_type(16)));
typedef float    v8f   __attribute__((ext_vector_type(8)));
typedef float    f4_t  __attribute__((ext_vector_type(4)));
typedef unsigned int u32x4 __attribute__((ext_vector_type(4)));
typedef int          i32x8 __attribute__((ext_vector_type(8)));
typedef int          i32x4 __attribute__((ext_vector_type(4)));

// ---------------------------------------------------------------------------
// WMMA wrapper: D = A(16x32 f16) * B(32x16 f16) + C(16x16 f32)
// ---------------------------------------------------------------------------
__device__ __forceinline__ v8f wmma_f16(v16h a, v16h b, v8f c) {
  return __builtin_amdgcn_wmma_f32_16x16x32_f16(
      /*neg_a=*/false, a, /*neg_b=*/false, b,
      /*c_mod=*/(short)0, c, /*reuse_a=*/false, /*reuse_b=*/false);
}

// ---------------------------------------------------------------------------
// Fragment loaders (CDNA5 ISA 7.12.2 layouts, wave32).
// A 16x32 (MxK): lane row = lane&15; lanes<16 hold K {kb+0..7, kb+16..23},
//                lanes>=16 hold K {kb+8..15, kb+24..31}.
// B 32x16 (KxN): lane col = lane&15; lanes<16 hold K {kb+0..7, kb+16..23},
//                lanes>=16 hold K {kb+8..15, kb+24..31}.
// `row` points at the start of this lane's row/column (row-major, K contiguous)
// ---------------------------------------------------------------------------
__device__ __forceinline__ v16h load_a_f16(const _Float16* row, int kb, int lhi) {
  const _Float16* p = row + kb + lhi * 8;
  h8_t lo = *(const h8_t*)(p);
  h8_t hi = *(const h8_t*)(p + 16);
  v16h r;
#pragma unroll
  for (int i = 0; i < 8; ++i) { r[i] = lo[i]; r[8 + i] = hi[i]; }
  return r;
}

__device__ __forceinline__ v16h load_b_f16(const _Float16* row, int kb, int lhi) {
  const _Float16* p = row + kb + lhi * 8;
  h8_t lo = *(const h8_t*)(p);
  h8_t hi = *(const h8_t*)(p + 16);
  v16h r;
#pragma unroll
  for (int i = 0; i < 8; ++i) { r[i] = lo[i]; r[8 + i] = hi[i]; }
  return r;
}

// ---------------------------------------------------------------------------
// TDM: 2D f16 tile (rows x row_elems) global -> LDS, one descriptor.
// D# per CDNA5 ISA 8.3/8.4. clang-23 6-arg builtin form:
//   (u32x4 g0, i32x8 g1, i32x4 g2, i32x4 g3, i32x8 extra, i32 cpol)
// ---------------------------------------------------------------------------
__device__ __forceinline__ void tdm_load_2d_f16_to_lds(
    unsigned lds_off, const void* gptr, unsigned rows, unsigned row_elems,
    unsigned row_stride_elems) {
  const unsigned long long ga = (unsigned long long)(uintptr_t)gptr;
  u32x4 g0;
  g0[0] = 1u;                                   // count=1 (valid user D#)
  g0[1] = lds_off;                              // lds_addr (bytes)
  g0[2] = (unsigned)(ga & 0xFFFFFFFFu);         // global_addr[31:0]
  g0[3] = (unsigned)((ga >> 32) & 0x01FFFFFFu)  // global_addr[56:32]
          | (2u << 30);                         // type=2 ("image")
  i32x8 g1;
  g1[0] = (int)(1u << 16);                      // data_size=1 -> 2 bytes
  g1[1] = (int)((row_elems & 0xFFFFu) << 16);   // tensor_dim0[15:0]
  g1[2] = (int)((row_elems >> 16) | (rows << 16));        // dim0 hi | dim1 lo
  g1[3] = (int)((row_elems & 0xFFFFu) << 16);   // tile_dim0 (dim1 hi = 0)
  g1[4] = (int)rows;                            // tile_dim1 (tile_dim2=0)
  g1[5] = (int)row_stride_elems;                // tensor_dim0_stride[31:0]
  g1[6] = 0;                                    // stride0 hi | stride1 lo
  g1[7] = 0;
  i32x4 z4 = {0, 0, 0, 0};                      // groups 2/3 unused (2D)
  i32x8 z8 = {0, 0, 0, 0, 0, 0, 0, 0};
  __builtin_amdgcn_tensor_load_to_lds(g0, g1, z4, z4, z8, 0);
}

// ---------------------------------------------------------------------------
// Kernel 0: bulk f32 -> f16 conversion (8 elements per thread, 16B stores)
// ---------------------------------------------------------------------------
__global__ __launch_bounds__(256) void cvt_f16_kernel(
    const float* __restrict__ src, _Float16* __restrict__ dst) {
  const size_t i = ((size_t)blockIdx.x * 256 + threadIdx.x) * 8;
  f4_t a = *(const f4_t*)(src + i);
  f4_t b = *(const f4_t*)(src + i + 4);
  h8_t o;
#pragma unroll
  for (int j = 0; j < 4; ++j) {
    o[j]     = (_Float16)a[j];
    o[4 + j] = (_Float16)b[j];
  }
  *(h8_t*)(dst + i) = o;
}

// ---------------------------------------------------------------------------
// Kernel 1: linear projection  dst = feat @ w^T + bias  (f16 in, f16 out)
// Block = 256 threads (8 wave32s, 4x2), tile 128(M) x 128(N), K stepped by 32.
// Each wave: 32x64 output via 2x4 WMMA register blocking (f32 accum).
// store_tr==0: dst[m][d] row-major f16 (Q, K)
// store_tr==1: dst transposed per batch: vt[b][d][n] (V), packed h8 stores
//              straight out of the C/D accumulator layout.
// ---------------------------------------------------------------------------
__global__ __launch_bounds__(256) void qkv_gemm_kernel(
    const _Float16* __restrict__ feat, const _Float16* __restrict__ w,
    const float* __restrict__ bias, _Float16* __restrict__ dst, int store_tr) {
  const int tid    = threadIdx.x;
  const int lane   = tid & 31;
  const int lane15 = lane & 15;
  const int lhi    = lane >> 4;
  const int wid    = tid >> 5;
  const int m0     = blockIdx.x * 128 + (wid >> 1) * 32; // wave M base
  const int n0     = blockIdx.y * 128 + (wid & 1) * 64;  // wave N base

  const _Float16* arow0 = feat + (size_t)(m0 + lane15) * C_DIM;
  const _Float16* arow1 = arow0 + (size_t)16 * C_DIM;
  const _Float16* brow[4];
#pragma unroll
  for (int j = 0; j < 4; ++j)
    brow[j] = w + (size_t)(n0 + j * 16 + lane15) * C_DIM;

  v8f acc[2][4] = {};

  for (int kc = 0; kc < C_DIM; kc += 32) {
    v16h a0 = load_a_f16(arow0, kc, lhi);
    v16h a1 = load_a_f16(arow1, kc, lhi);
#pragma unroll
    for (int j = 0; j < 4; ++j) {
      v16h b = load_b_f16(brow[j], kc, lhi);
      acc[0][j] = wmma_f16(a0, b, acc[0][j]);
      acc[1][j] = wmma_f16(a1, b, acc[1][j]);
    }
  }

#pragma unroll
  for (int j = 0; j < 4; ++j) {
    const int d    = n0 + j * 16 + lane15; // C/D layout: lane holds fixed col
    const float bj = bias[d];
#pragma unroll
    for (int i = 0; i < 2; ++i) {
      if (!store_tr) {
#pragma unroll
        for (int r = 0; r < 8; ++r) {
          const int m = m0 + i * 16 + 8 * lhi + r;
          dst[(size_t)m * C_DIM + d] = (_Float16)(acc[i][j][r] + bj);
        }
      } else {
        const int mb   = m0 + i * 16 + 8 * lhi; // 8-aligned token base
        const int bIdx = mb >> 11;              // token / 2048
        const int nIdx = mb & (N_TOK - 1);
        h8_t pk;
#pragma unroll
        for (int r = 0; r < 8; ++r) pk[r] = (_Float16)(acc[i][j][r] + bj);
        *(h8_t*)(dst + ((size_t)bIdx * C_DIM + d) * N_TOK + nIdx) = pk;
      }
    }
  }
}

// ---------------------------------------------------------------------------
// Kernel 2: flash attention, one block per (16 query rows, batch).
// Q tile staged to LDS via the Tensor Data Mover (TENSORcnt). 8 waves: all
// compute scores (wave w -> keys kb+16w..), wave 0 runs the online softmax
// (m, l, corr, P in LDS), then wave w accumulates d-slice [64w, 64w+64) of O
// via P@V WMMAs. K rows and V^T rows are read straight from global f16
// (q/k/vt = 48 MB total -> resident in the 192 MB L2).
// ---------------------------------------------------------------------------
#define SS_PITCH 132 // f32 score row pitch (pad vs bank conflicts)
#define SP_PITCH 136 // f16 prob row pitch (16B aligned rows)

__global__ __launch_bounds__(256) void flash_attn_kernel(
    const _Float16* __restrict__ qh, const _Float16* __restrict__ kh,
    const _Float16* __restrict__ vt, const float* __restrict__ feat,
    float* __restrict__ out) {
  __shared__ _Float16 sQ[16 * C_DIM];     // 16 KB: query tile
  __shared__ float    sS[16 * SS_PITCH];  // score tile (f32)
  __shared__ _Float16 sP[16 * SP_PITCH];  // prob tile (f16, A-frag source)
  __shared__ float    s_m[16], s_l[16], s_corr[16];

  const int tid    = threadIdx.x;
  const int lane   = tid & 31;
  const int lane15 = lane & 15;
  const int lhi    = lane >> 4;
  const int wid    = tid >> 5;
  const int batch  = blockIdx.y;
  const size_t qrow0 = (size_t)batch * N_TOK + blockIdx.x * 16;

  // ---- stage Q tile (16 x 512 f16) with one TDM descriptor (wave 0 issues)
  if (wid == 0) {
    tdm_load_2d_f16_to_lds((unsigned)(uintptr_t)(void*)sQ,
                           qh + qrow0 * C_DIM,
                           /*rows=*/16, /*row_elems=*/C_DIM,
                           /*row_stride=*/C_DIM);
    __builtin_amdgcn_s_wait_tensorcnt(0);
  }
  if (tid < 16) { s_m[tid] = -1e30f; s_l[tid] = 0.0f; }
  __syncthreads();

  v8f acc[4] = {};
  const int dbase = wid * 64;

  for (int kb = 0; kb < N_TOK; kb += 128) {
    // ---- phase 1: scores S[16 x 128]; wave w owns keys kb+16w..kb+16w+15
    v8f sc = {};
    const _Float16* krow =
        kh + ((size_t)batch * N_TOK + kb + wid * 16 + lane15) * C_DIM;
    if (kb + 128 < N_TOK) // warm next key block (L2-resident -> WGP$)
      __builtin_prefetch(krow + (size_t)128 * C_DIM, 0, 1);
#pragma unroll
    for (int kc = 0; kc < C_DIM; kc += 32) {
      v16h a = load_a_f16(sQ + lane15 * C_DIM, kc, lhi);
      v16h b = load_b_f16(krow, kc, lhi);
      sc = wmma_f16(a, b, sc);
    }
    {
      const int col = wid * 16 + lane15;
#pragma unroll
      for (int r = 0; r < 8; ++r) sS[(r + 8 * lhi) * SS_PITCH + col] = sc[r];
    }
    __syncthreads();

    // ---- phase 2: online softmax row update (wave 0, lane = row)
    if (wid == 0 && lane < 16) {
      const int row    = lane;
      const float mold = s_m[row];
      float mx = mold;
      for (int j = 0; j < 128; ++j) mx = fmaxf(mx, sS[row * SS_PITCH + j]);
      const float corr = __expf(mold - mx);
      float l = s_l[row] * corr;
      for (int j = 0; j < 128; ++j) {
        const float e = __expf(sS[row * SS_PITCH + j] - mx);
        l += e;
        sP[row * SP_PITCH + j] = (_Float16)e;
      }
      s_m[row] = mx; s_l[row] = l; s_corr[row] = corr;
    }
    __syncthreads();

    // ---- phase 3: rescale O, then O += P @ V (wave w: d slice 64w..64w+63)
#pragma unroll
    for (int r = 0; r < 8; ++r) {
      const float c = s_corr[r + 8 * lhi];
#pragma unroll
      for (int j = 0; j < 4; ++j) acc[j][r] *= c;
    }
#pragma unroll
    for (int kk = 0; kk < 128; kk += 32) {
      v16h a = load_a_f16(sP + lane15 * SP_PITCH, kk, lhi);
#pragma unroll
      for (int j = 0; j < 4; ++j) {
        const _Float16* vrow =
            vt + ((size_t)batch * C_DIM + dbase + j * 16 + lane15) * N_TOK + kb;
        v16h b = load_b_f16(vrow, kk, lhi);
        acc[j] = wmma_f16(a, b, acc[j]);
      }
    }
    __syncthreads();
  }

  // ---- epilogue: out = feat + (O / l) * (1/sqrt(C))  [post-softmax scale]
  const float inv_sqrt_c = 0.044194173824159216f; // 1/sqrt(512)
#pragma unroll
  for (int j = 0; j < 4; ++j) {
    const int d = dbase + j * 16 + lane15;
#pragma unroll
    for (int r = 0; r < 8; ++r) {
      const int row  = r + 8 * lhi;
      const size_t g = (qrow0 + row) * C_DIM + d;
      out[g] = feat[g] + acc[j][r] * (inv_sqrt_c / s_l[row]);
    }
  }
}

// ---------------------------------------------------------------------------
// Host launcher
// ---------------------------------------------------------------------------
extern "C" void kernel_launch(void* const* d_in, const int* in_sizes, int n_in,
                              void* d_out, int out_size, void* d_ws,
                              size_t ws_size, hipStream_t stream) {
  const float* feat = (const float*)d_in[0];
  const float* wq   = (const float*)d_in[1];
  const float* bq   = (const float*)d_in[2];
  const float* wk   = (const float*)d_in[3];
  const float* bk   = (const float*)d_in[4];
  const float* wv   = (const float*)d_in[5];
  const float* bv   = (const float*)d_in[6];
  float* out = (float*)d_out;

  // workspace: q | k | v^T | feat(f16) | wq(f16) | wk(f16) | wv(f16)
  const size_t MC = (size_t)M_TOT * C_DIM; // 8M elements
  const size_t CC = (size_t)C_DIM * C_DIM; // 256K elements
  _Float16* qh    = (_Float16*)d_ws;
  _Float16* kh    = qh + MC;
  _Float16* vt    = kh + MC;
  _Float16* feath = vt + MC;
  _Float16* wqh   = feath + MC;
  _Float16* wkh   = wqh + CC;
  _Float16* wvh   = wkh + CC;

  // 0) one-time f32 -> f16 conversions (feature reused 12x, weights 128x)
  cvt_f16_kernel<<<(int)(MC / 2048), 256, 0, stream>>>(feat, feath);
  cvt_f16_kernel<<<(int)(CC / 2048), 256, 0, stream>>>(wq, wqh);
  cvt_f16_kernel<<<(int)(CC / 2048), 256, 0, stream>>>(wk, wkh);
  cvt_f16_kernel<<<(int)(CC / 2048), 256, 0, stream>>>(wv, wvh);

  // 1) QKV projections (V stored transposed)
  dim3 g1(M_TOT / 128, C_DIM / 128); // 128 x 4 blocks
  qkv_gemm_kernel<<<g1, 256, 0, stream>>>(feath, wqh, bq, qh, 0);
  qkv_gemm_kernel<<<g1, 256, 0, stream>>>(feath, wkh, bk, kh, 0);
  qkv_gemm_kernel<<<g1, 256, 0, stream>>>(feath, wvh, bv, vt, 1);

  // 2) fused flash attention + residual
  dim3 g2(N_TOK / 16, B_DIM); // 128 x 8 blocks
  flash_attn_kernel<<<g2, 256, 0, stream>>>(qh, kh, vt, feat, out);
}
